// S4Disc_44796508897833
// MI455X (gfx1250) — matrix operand order repaired and checked by reading, not compiled, not run on hardware
//
#include <hip/hip_runtime.h>
#include <math.h>

// Problem dims (match reference)
#define BB 32
#define LLEN 4096
#define HH 256
#define NN2 32
#define NLAY 4
#define LN_EPS 1e-5f

typedef __attribute__((ext_vector_type(16))) __bf16 bf16x16;
typedef __attribute__((ext_vector_type(8)))  float  f32x8;
typedef __attribute__((ext_vector_type(8)))  unsigned short u16x8;

union BFrag { u16x8 h[2]; bf16x16 v; };

__device__ __forceinline__ unsigned short f2bf(float f) {
  unsigned u = __builtin_bit_cast(unsigned, f);
  unsigned r = (u + 0x7FFFu + ((u >> 16) & 1u)) >> 16;  // round-to-nearest-even
  return (unsigned short)r;
}

// ---------------------------------------------------------------------------
// Encoder: h[b,h,l] = x[b,l,0] * enc_w[h,0] + enc_b[h]   (CH == 1)
// hbuf layout: (B, H, L) f32
// ---------------------------------------------------------------------------
__global__ void k_encode(const float* __restrict__ x, const float* __restrict__ ew,
                         const float* __restrict__ eb, float* __restrict__ hbuf) {
  size_t idx = (size_t)blockIdx.x * blockDim.x + threadIdx.x;   // over B*H*L
  int l  = (int)(idx & (LLEN - 1));
  int hh = (int)((idx >> 12) & (HH - 1));
  int b  = (int)(idx >> 20);
  float xv = x[(size_t)b * LLEN + l];
  hbuf[idx] = xv * ew[hh] + eb[hh];
}

// ---------------------------------------------------------------------------
// Convert GLU weights (NL, 2H, H) f32 -> bf16 once
// ---------------------------------------------------------------------------
__global__ void k_wcvt(const float* __restrict__ w, unsigned short* __restrict__ wbf, int n) {
  int i = blockIdx.x * blockDim.x + threadIdx.x;
  if (i < n) wbf[i] = f2bf(w[i]);
}

// ---------------------------------------------------------------------------
// S4D diagonal-SSM scan + D skip + exact GELU, one wave32 per (b,h) channel,
// one lane per state mode (N2 == 32). Output written bf16 into yact laid out
// (B*L, H) row-major so the GEMM B-fragment load is two contiguous 16B reads.
// ---------------------------------------------------------------------------
__global__ void __launch_bounds__(256) k_scan(
    const float* __restrict__ hbuf, const float* __restrict__ log_dt,
    const float* __restrict__ log_A_real, const float* __restrict__ A_imag,
    const float* __restrict__ C_re, const float* __restrict__ C_im,
    const float* __restrict__ Dp, unsigned short* __restrict__ yact, int layer) {
  int lane = threadIdx.x & 31;
  int wid  = threadIdx.x >> 5;
  int ch = blockIdx.x * 8 + wid;          // channel = b*H + hh
  int b = ch >> 8, hh = ch & (HH - 1);
  int pi = layer * HH + hh;
  int ni = pi * NN2 + lane;

  // Discretize: w = exp(dt*A), Ck = C*(w-1)/A
  float dt  = expf(log_dt[pi]);
  float Are = -expf(log_A_real[ni]);
  float Aim = A_imag[ni];
  float dre = Are * dt, dim = Aim * dt;
  float er  = expf(dre);
  float wre = er * cosf(dim), wim = er * sinf(dim);
  float e1r = wre - 1.f, e1i = wim;
  float Cre = C_re[ni], Cim = C_im[ni];
  float numr = Cre * e1r - Cim * e1i;
  float numi = Cre * e1i + Cim * e1r;
  float inv  = 1.f / (Are * Are + Aim * Aim);
  float Ckr = (numr * Are + numi * Aim) * inv;
  float Cki = (numi * Are - numr * Aim) * inv;
  float Dv  = Dp[pi];

  const float* u = hbuf + ((size_t)b * HH + hh) * LLEN;
  unsigned short* yo = yact + (size_t)b * LLEN * HH + hh;

  float sr = 0.f, si = 0.f;
  for (int l0 = 0; l0 < LLEN; l0 += 32) {
    float uv = u[l0 + lane];          // coalesced block load, broadcast below
    float ykeep = 0.f;
    for (int j = 0; j < 32; ++j) {
      float uu = __shfl(uv, j, 32);
      float nsr = wre * sr - wim * si + uu;   // s = w*s + u
      float nsi = wre * si + wim * sr;
      sr = nsr; si = nsi;
      float p = Ckr * sr - Cki * si;          // Re(Ck * s)
      #pragma unroll
      for (int off = 16; off; off >>= 1) p += __shfl_xor(p, off, 32);
      float y  = 2.f * p + Dv * uu;
      float gl = 0.5f * y * (1.f + erff(y * 0.70710678118654752f));  // exact GELU
      if (lane == j) ykeep = gl;
    }
    yo[(size_t)(l0 + lane) * HH] = f2bf(ykeep);
  }
}

// ---------------------------------------------------------------------------
// GLU 1x1 conv as bf16 WMMA GEMM: z = W(2H x H) * Y(H x B*L), fused
// GLU (a * sigmoid(g)) + bias + residual add into hbuf.
//
// Block = 8 waves; block owns TWO 16-row pair-tiles (32 a-rows + the matching
// 32 g-rows, halving B-matrix L2 traffic vs 16-row blocks), weights staged in
// LDS via CDNA5 async memory->LDS loads (ASYNCcnt path); each wave owns one
// 16-col tile and runs 4 WMMAs per B fragment.
// Fragment layouts per ISA 7.12.2 (wave32).
// ---------------------------------------------------------------------------
__global__ void __launch_bounds__(256) k_glu_gemm(
    const unsigned short* __restrict__ yact, const unsigned short* __restrict__ wbf,
    const float* __restrict__ glu_b, float* __restrict__ hbuf, int layer) {
  __shared__ unsigned short wlds[2][32][HH];   // [a|g][row][k], 32 KB

  int t = threadIdx.x;
  int lane = t & 31, wid = t >> 5;
  int rt2 = blockIdx.x >> 10;                  // 8 row groups of 32 a-rows
  int cg  = blockIdx.x & 1023;
  int ct  = cg * 8 + wid;                      // 8192 col tiles

  // --- Async weight staging: 2048 x 16B moves, 8 per thread, direct to LDS
  const unsigned short* wl = wbf + (size_t)layer * 2 * HH * HH;
  unsigned ldsbase = (unsigned)(unsigned long long)&wlds[0][0][0];
  #pragma unroll
  for (int i = 0; i < 8; ++i) {
    int idx   = i * 256 + t;                   // 16B unit index (0..2047)
    int row   = idx >> 5;                      // 0..63 (32 a-rows then 32 g-rows)
    int c8    = idx & 31;                      // 16B unit within row
    int which = row >> 5;                      // 0 = a, 1 = g
    int r32   = row & 31;
    unsigned long long src = (unsigned long long)(const void*)(
        wl + (size_t)(which * HH + rt2 * 32 + r32) * HH + c8 * 8);
    unsigned dst = ldsbase + (unsigned)(((which * 32 + r32) * HH + c8 * 8) * 2);
    asm volatile("global_load_async_to_lds_b128 %0, %1, off"
                 :: "v"(dst), "v"(src) : "memory");
  }
  asm volatile("s_wait_asynccnt 0x0" ::: "memory");
  __syncthreads();

  f32x8 ca0 = {0.f, 0.f, 0.f, 0.f, 0.f, 0.f, 0.f, 0.f};
  f32x8 ca1 = ca0, cg0 = ca0, cg1 = ca0;

  int m     = lane & 15;                 // A: lanes 0-15 and 16-31 both hold M=0..15
  int kbase = (lane >> 4) * 8;           // A: upper half-wave holds K+8 slice
  int kb    = (lane >> 4) * 8;           // B: upper half-wave holds K+8 slice
  int nn    = ct * 16 + (lane & 15);     // B column (global position b*L+l)
  const unsigned short* yrow = yact + (size_t)nn * HH;

  #pragma unroll
  for (int ks = 0; ks < 8; ++ks) {       // K = 256 in steps of 32
    int k0 = ks * 32;
    BFrag a0, a1, g0, g1, bf;
    a0.h[0] = *reinterpret_cast<const u16x8*>(&wlds[0][m][k0 + kbase]);
    a0.h[1] = *reinterpret_cast<const u16x8*>(&wlds[0][m][k0 + 16 + kbase]);
    a1.h[0] = *reinterpret_cast<const u16x8*>(&wlds[0][16 + m][k0 + kbase]);
    a1.h[1] = *reinterpret_cast<const u16x8*>(&wlds[0][16 + m][k0 + 16 + kbase]);
    g0.h[0] = *reinterpret_cast<const u16x8*>(&wlds[1][m][k0 + kbase]);
    g0.h[1] = *reinterpret_cast<const u16x8*>(&wlds[1][m][k0 + 16 + kbase]);
    g1.h[0] = *reinterpret_cast<const u16x8*>(&wlds[1][16 + m][k0 + kbase]);
    g1.h[1] = *reinterpret_cast<const u16x8*>(&wlds[1][16 + m][k0 + 16 + kbase]);
    const u16x8* yp = reinterpret_cast<const u16x8*>(yrow + k0 + kb);
    bf.h[0] = yp[0];
    bf.h[1] = yp[2];
    if (ks < 7) __builtin_prefetch(yrow + (ks + 1) * 32 + kb, 0, 0);
    ca0 = __builtin_amdgcn_wmma_f32_16x16x32_bf16(false, a0.v, false, bf.v,
                                                  (short)0, ca0, false, false);
    ca1 = __builtin_amdgcn_wmma_f32_16x16x32_bf16(false, a1.v, false, bf.v,
                                                  (short)0, ca1, false, false);
    cg0 = __builtin_amdgcn_wmma_f32_16x16x32_bf16(false, g0.v, false, bf.v,
                                                  (short)0, cg0, false, false);
    cg1 = __builtin_amdgcn_wmma_f32_16x16x32_bf16(false, g1.v, false, bf.v,
                                                  (short)0, cg1, false, false);
  }

  // Epilogue: bias + GLU + residual add into hbuf (B,H,L)
  int moff = (lane >> 4) * 8;            // C/D: m = r + 8*(lane>=16)
  long ncol = (long)ct * 16 + (lane & 15);
  int b = (int)(ncol >> 12);
  int l = (int)(ncol & (LLEN - 1));
  #pragma unroll
  for (int sub = 0; sub < 2; ++sub) {
    #pragma unroll
    for (int r = 0; r < 8; ++r) {
      int mg = rt2 * 32 + sub * 16 + r + moff;
      float av = (sub ? ca1[r] : ca0[r]) + glu_b[layer * 2 * HH + mg];
      float gv = (sub ? cg1[r] : cg0[r]) + glu_b[layer * 2 * HH + HH + mg];
      float glu = av * (1.f / (1.f + expf(-gv)));
      hbuf[((size_t)b * HH + mg) * LLEN + l] += glu;
    }
  }
}

// ---------------------------------------------------------------------------
// Post-norm LayerNorm over H, one wave per (b,l) position (lane holds 8 ch)
// ---------------------------------------------------------------------------
__global__ void __launch_bounds__(256) k_ln(float* __restrict__ hbuf,
                                            const float* __restrict__ ln_g,
                                            const float* __restrict__ ln_b, int layer) {
  int lane = threadIdx.x & 31, wid = threadIdx.x >> 5;
  long pos = (long)blockIdx.x * 8 + wid;       // b*L + l
  int b = (int)(pos >> 12), l = (int)(pos & (LLEN - 1));
  float v[8], s = 0.f, s2 = 0.f;
  #pragma unroll
  for (int q = 0; q < 8; ++q) {
    int chn = lane * 8 + q;
    v[q] = hbuf[((size_t)b * HH + chn) * LLEN + l];
    s += v[q]; s2 += v[q] * v[q];
  }
  #pragma unroll
  for (int off = 16; off; off >>= 1) {
    s  += __shfl_xor(s, off, 32);
    s2 += __shfl_xor(s2, off, 32);
  }
  float mu   = s * (1.f / HH);
  float var  = s2 * (1.f / HH) - mu * mu;      // biased var (matches jnp.var)
  float rinv = rsqrtf(var + LN_EPS);
  #pragma unroll
  for (int q = 0; q < 8; ++q) {
    int chn = lane * 8 + q;
    float o = (v[q] - mu) * rinv * ln_g[layer * HH + chn] + ln_b[layer * HH + chn];
    hbuf[((size_t)b * HH + chn) * LLEN + l] = o;
  }
}

// ---------------------------------------------------------------------------
// Decoder stage 1: y1[b,h] = <h[b,h,:], dec1_w> + dec1_b   (wave per channel)
// ---------------------------------------------------------------------------
__global__ void k_dec1(const float* __restrict__ hbuf, const float* __restrict__ w1,
                       const float* __restrict__ b1, float* __restrict__ y1) {
  int lane = threadIdx.x & 31, wid = threadIdx.x >> 5;
  int ch = blockIdx.x * 8 + wid;               // b*H + hh
  const float* row = hbuf + (size_t)ch * LLEN;
  float acc = 0.f;
  for (int l = lane; l < LLEN; l += 32) acc += row[l] * w1[l];
  #pragma unroll
  for (int off = 16; off; off >>= 1) acc += __shfl_xor(acc, off, 32);
  if (lane == 0) y1[ch] = acc + b1[0];
}

// ---------------------------------------------------------------------------
// Decoder stage 2: out[b] = sigmoid(<y1[b,:], dec2_w> + dec2_b)
// ---------------------------------------------------------------------------
__global__ void k_dec2(const float* __restrict__ y1, const float* __restrict__ w2,
                       const float* __restrict__ b2, float* __restrict__ out) {
  int b = blockIdx.x, lane = threadIdx.x;
  float acc = 0.f;
  for (int hh = lane; hh < HH; hh += 32) acc += y1[b * HH + hh] * w2[hh];
  #pragma unroll
  for (int off = 16; off; off >>= 1) acc += __shfl_xor(acc, off, 32);
  if (lane == 0) out[b] = 1.f / (1.f + expf(-(acc + b2[0])));
}

// ---------------------------------------------------------------------------
extern "C" void kernel_launch(void* const* d_in, const int* in_sizes, int n_in,
                              void* d_out, int out_size, void* d_ws, size_t ws_size,
                              hipStream_t stream) {
  const float* x          = (const float*)d_in[0];
  const float* enc_w      = (const float*)d_in[1];
  const float* enc_b      = (const float*)d_in[2];
  const float* log_dt     = (const float*)d_in[3];
  const float* log_A_real = (const float*)d_in[4];
  const float* A_imag     = (const float*)d_in[5];
  const float* C_re       = (const float*)d_in[6];
  const float* C_im       = (const float*)d_in[7];
  const float* Dp         = (const float*)d_in[8];
  const float* glu_w      = (const float*)d_in[9];
  const float* glu_b      = (const float*)d_in[10];
  const float* ln_g       = (const float*)d_in[11];
  const float* ln_b       = (const float*)d_in[12];
  const float* dec1_w     = (const float*)d_in[13];
  const float* dec1_b     = (const float*)d_in[14];
  const float* dec2_w     = (const float*)d_in[15];
  const float* dec2_b     = (const float*)d_in[16];

  char* ws = (char*)d_ws;
  size_t off = 0;
  float* hbuf = (float*)(ws + off);           off += (size_t)BB * HH * LLEN * 4;  // 128 MB
  unsigned short* yact = (unsigned short*)(ws + off); off += (size_t)BB * LLEN * HH * 2; // 64 MB
  unsigned short* wbf  = (unsigned short*)(ws + off); off += (size_t)NLAY * 2 * HH * HH * 2; // 1 MB
  float* y1   = (float*)(ws + off);

  // Encoder + weight conversion
  k_encode<<<(BB * HH * LLEN) / 256, 256, 0, stream>>>(x, enc_w, enc_b, hbuf);
  k_wcvt<<<(NLAY * 2 * HH * HH) / 256, 256, 0, stream>>>(glu_w, wbf, NLAY * 2 * HH * HH);

  // Layer stack
  for (int layer = 0; layer < NLAY; ++layer) {
    k_scan<<<(BB * HH) / 8, 256, 0, stream>>>(hbuf, log_dt, log_A_real, A_imag,
                                              C_re, C_im, Dp, yact, layer);
    k_glu_gemm<<<8 * 1024, 256, 0, stream>>>(yact, wbf, glu_b, hbuf, layer);
    k_ln<<<(BB * LLEN) / 8, 256, 0, stream>>>(hbuf, ln_g, ln_b, layer);
  }

  // Decoder
  k_dec1<<<(BB * HH) / 8, 256, 0, stream>>>(hbuf, dec1_w, dec1_b, y1);
  k_dec2<<<BB, 32, 0, stream>>>(y1, dec2_w, dec2_b, (float*)d_out);
}
